// multi_head_enc_dec_attn_75265006895683
// MI455X (gfx1250) — matrix operand order repaired and checked by reading, not compiled, not run on hardware
//
#include <hip/hip_runtime.h>
#include <hip/hip_bf16.h>
#include <math.h>
#include <stdint.h>

typedef __bf16 bf16;
typedef __attribute__((ext_vector_type(16))) __bf16 v16bf;
typedef __attribute__((ext_vector_type(8)))  float  v8f;
typedef __attribute__((ext_vector_type(4)))  unsigned int u32x4;
typedef __attribute__((ext_vector_type(8)))  int i32x8;
typedef __attribute__((ext_vector_type(4)))  int i32x4;

#define EMB   1024
#define HD    64
#define NH    16
#define SEQ   2048
#define BATCH 2
#define MROWS (BATCH * SEQ)
#define LN_EPSF 1e-5f

union AFrag { v16bf v; bf16 e[16]; };
union CFrag { v8f  v; float f[8]; };

static __device__ __forceinline__ v8f wmma_bf16(v16bf a, v16bf b, v8f c) {
  // D = A(16x32 bf16) * B(32x16 bf16) + C(16x16 f32)
  return __builtin_amdgcn_wmma_f32_16x16x32_bf16(
      /*neg_a=*/false, a, /*neg_b=*/false, b,
      /*c_mod=*/(short)0, c, /*reuse_a=*/false, /*reuse_b=*/false);
}

// ---------------------------------------------------------------------------
// Tensor Data Mover: load a 2D tile (tile_d1 rows x tile_d0 contiguous elems,
// 2-byte data) of a row-major tensor into LDS.  Probe-verified 5-arg builtin
// on ROCm 7.2 (clang-22); 6-arg variant on clang-23+.  Fallback: none here --
// callers guard with HAVE_TDM.
// ---------------------------------------------------------------------------
#if defined(__HIP_DEVICE_COMPILE__) && __has_builtin(__builtin_amdgcn_tensor_load_to_lds)
#define HAVE_TDM 1
#else
#define HAVE_TDM 0
#endif

#if HAVE_TDM
static __device__ __forceinline__ void tdm_load_2d_bf16(
    unsigned lds_addr, const bf16* gptr,
    unsigned tensor_d0, unsigned tensor_d1,
    unsigned tile_d0, unsigned tile_d1, unsigned stride0)
{
  const unsigned long long ga = (unsigned long long)(uintptr_t)gptr;
  // D# group 0: count=1 | lds_addr | global_addr[56:0] | type=2 ("image")
  const u32x4 g0 = {
      1u,
      lds_addr,
      (unsigned)(ga & 0xffffffffu),
      (unsigned)((ga >> 32) & 0x01ffffffu) | (2u << 30)
  };
  // D# group 1: data_size=2B, tensor_dim0/1, tile_dim0/1, dim0 stride
  const i32x8 g1 = {
      (int)(1u << 16),                                        // data_size code 1
      (int)((tensor_d0 & 0xffffu) << 16),                     // dim0[15:0]
      (int)((tensor_d0 >> 16) | ((tensor_d1 & 0xffffu) << 16)),
      (int)((tensor_d1 >> 16) | ((tile_d0 & 0xffffu) << 16)),
      (int)(tile_d1 & 0xffffu),                               // tile_dim1, tile_dim2=0
      (int)stride0,                                           // dim0 stride [47:0]
      0, 0
  };
  const i32x4 z4 = {0, 0, 0, 0};
#if defined(__clang_major__) && (__clang_major__ >= 23)
  const i32x8 z8 = {0, 0, 0, 0, 0, 0, 0, 0};
  __builtin_amdgcn_tensor_load_to_lds(g0, g1, z4, z4, z8, 0);
#else
  __builtin_amdgcn_tensor_load_to_lds(g0, g1, z4, z4, 0);
#endif
}
#endif

// ---------------------------------------------------------------------------
// Kernel 1: Q/K/V projection.  GEMM  X(4096x1024 f32) * W(1024x64 per head)
// blockIdx.z selects {Q from dec, K from enc, V from enc}.
// Block tile 128(M) x 64(N); 8 waves 4(M) x 2(N); each wave 2x2 16x16 accums.
// Output layout: [b, h, s, d] bf16.
// ---------------------------------------------------------------------------
__global__ __launch_bounds__(256) void qkv_proj_kernel(
    const float* __restrict__ enc, const float* __restrict__ dec,
    const float* __restrict__ Wq, const float* __restrict__ Wk,
    const float* __restrict__ Wv,
    bf16* __restrict__ qo, bf16* __restrict__ ko, bf16* __restrict__ vo)
{
  const int which = blockIdx.z;
  const float* X = (which == 0) ? dec : enc;
  const float* W = (which == 0) ? Wq : ((which == 1) ? Wk : Wv);
  bf16* O        = (which == 0) ? qo : ((which == 1) ? ko : vo);

  const int n0 = blockIdx.x * 64;
  const int m0 = blockIdx.y * 128;
  const int h  = n0 >> 6;

  __shared__ bf16 As[128][34];
  __shared__ bf16 Bs[32][66];

  const int tid  = threadIdx.x;
  const int lane = tid & 31;
  const int wave = tid >> 5;
  const int wm   = wave & 3;
  const int wn   = wave >> 2;
  const int half = lane >> 4;
  const int l15  = lane & 15;

  CFrag c[2][2];
  #pragma unroll
  for (int mi = 0; mi < 2; ++mi)
    #pragma unroll
    for (int ni = 0; ni < 2; ++ni)
      #pragma unroll
      for (int r = 0; r < 8; ++r) c[mi][ni].f[r] = 0.0f;

  for (int k0 = 0; k0 < EMB; k0 += 32) {
    { // A tile: 16 contiguous f32 -> bf16 per thread
      const int row = tid >> 1, hc = (tid & 1) * 16;
      const float* ap = X + (size_t)(m0 + row) * EMB + k0 + hc;
      __builtin_prefetch(ap + 64, 0, 0);
      const float4* src = (const float4*)ap;
      #pragma unroll
      for (int j = 0; j < 4; ++j) {
        float4 f = src[j];
        const int base = hc + j * 4;
        As[row][base + 0] = (bf16)f.x;
        As[row][base + 1] = (bf16)f.y;
        As[row][base + 2] = (bf16)f.z;
        As[row][base + 3] = (bf16)f.w;
      }
    }
    { // B tile: W[h][e][d], 32 x 64
      const int kr = tid >> 3, cg = (tid & 7) * 8;
      const float4* src =
          (const float4*)(W + ((size_t)h * EMB + (k0 + kr)) * HD + cg);
      #pragma unroll
      for (int j = 0; j < 2; ++j) {
        float4 f = src[j];
        const int base = cg + j * 4;
        Bs[kr][base + 0] = (bf16)f.x;
        Bs[kr][base + 1] = (bf16)f.y;
        Bs[kr][base + 2] = (bf16)f.z;
        Bs[kr][base + 3] = (bf16)f.w;
      }
    }
    __syncthreads();

    AFrag a[2], b[2];
    const int kb = half * 8;   // A: K = ks + i
    const int ks = half * 16;  // B: K = ks + i
    #pragma unroll
    for (int mi = 0; mi < 2; ++mi) {
      const int row = wm * 32 + mi * 16 + l15;
      #pragma unroll
      for (int i = 0; i < 16; ++i)
        a[mi].e[i] = As[row][ks + i];
    }
    #pragma unroll
    for (int ni = 0; ni < 2; ++ni) {
      const int col = wn * 32 + ni * 16 + l15;
      #pragma unroll
      for (int i = 0; i < 16; ++i)
        b[ni].e[i] = Bs[ks + i][col];
    }
    #pragma unroll
    for (int mi = 0; mi < 2; ++mi)
      #pragma unroll
      for (int ni = 0; ni < 2; ++ni)
        c[mi][ni].v = wmma_bf16(a[mi].v, b[ni].v, c[mi][ni].v);
    __syncthreads();
  }

  #pragma unroll
  for (int mi = 0; mi < 2; ++mi)
    #pragma unroll
    for (int ni = 0; ni < 2; ++ni)
      #pragma unroll
      for (int r = 0; r < 8; ++r) {
        const int m  = m0 + wm * 32 + mi * 16 + r + half * 8;
        const int n  = n0 + wn * 32 + ni * 16 + l15;
        const int bb = m >> 11, s = m & (SEQ - 1), d = n & 63;
        O[(((size_t)bb * NH + h) * SEQ + s) * HD + d] = (bf16)c[mi][ni].f[r];
      }
}

// ---------------------------------------------------------------------------
// Kernel 2: flash attention per (b,h).  K/V tiles (32x64 bf16) are staged in
// LDS once per block -- via the Tensor Data Mover (double buffered, wave 0
// issues, TENSORcnt paced) when available, else cooperative 16B copies.
// Each wave owns 16 query rows: 4 WMMA scores, online softmax, P through
// per-wave LDS (C-layout -> A-layout), 4 WMMA ctx.
// ---------------------------------------------------------------------------
__global__ __launch_bounds__(256) void attn_kernel(
    const bf16* __restrict__ q, const bf16* __restrict__ k,
    const bf16* __restrict__ v, bf16* __restrict__ ctx)
{
  const int bh   = blockIdx.y;                 // b*NH + h
  const int tid  = threadIdx.x;
  const int wave = tid >> 5;
  const int lane = tid & 31;
  const int half = lane >> 4;
  const int l15  = lane & 15;
  const int qrow0 = blockIdx.x * 128 + wave * 16;

  const bf16* qp = q + ((size_t)bh * SEQ + qrow0) * HD;
  const bf16* kp = k + (size_t)bh * SEQ * HD;
  const bf16* vp = v + (size_t)bh * SEQ * HD;

  __shared__ bf16 Ks[2][32][HD];               // double-buffered K tile
  __shared__ bf16 Vs[2][32][HD];               // double-buffered V tile
  __shared__ bf16 Plds[8][16][33];             // per-wave P staging

  // Q fragments (d=0..31, d=32..63) resident for the whole key loop
  AFrag aq[2];
  {
    const int kb = half * 8;
    #pragma unroll
    for (int ci = 0; ci < 2; ++ci)
      #pragma unroll
      for (int i = 0; i < 16; ++i) {
        const int d = ci * 32 + half * 16 + i;
        aq[ci].e[i] = qp[(size_t)l15 * HD + d];
      }
  }

  CFrag acc[4];
  #pragma unroll
  for (int dt = 0; dt < 4; ++dt)
    #pragma unroll
    for (int r = 0; r < 8; ++r) acc[dt].f[r] = 0.0f;

  float mrun[8], lrun[8];
  #pragma unroll
  for (int r = 0; r < 8; ++r) { mrun[r] = -1e30f; lrun[r] = 0.0f; }

  auto stage = [&](int buf, int kt) {
#if HAVE_TDM
    if (wave == 0) {
      tdm_load_2d_bf16((unsigned)(uintptr_t)&Ks[buf][0][0],
                       kp + (size_t)kt * HD, HD, SEQ, HD, 32, HD);
      tdm_load_2d_bf16((unsigned)(uintptr_t)&Vs[buf][0][0],
                       vp + (size_t)kt * HD, HD, SEQ, HD, 32, HD);
    }
#else
    const int row = tid >> 3, c8 = (tid & 7) * 8;
    *(uint4*)&Ks[buf][row][c8] = *(const uint4*)(kp + (size_t)(kt + row) * HD + c8);
    *(uint4*)&Vs[buf][row][c8] = *(const uint4*)(vp + (size_t)(kt + row) * HD + c8);
#endif
  };

  stage(0, 0);

  const float scale = 0.125f;                  // 1/sqrt(64)
  const int NIT = SEQ / 32;

  for (int it = 0; it < NIT; ++it) {
    const int buf = it & 1;
    if (it + 1 < NIT) stage(buf ^ 1, (it + 1) * 32);
#if HAVE_TDM
    if (wave == 0) {
      if (it + 1 < NIT) __builtin_amdgcn_s_wait_tensorcnt((short)2);
      else              __builtin_amdgcn_s_wait_tensorcnt((short)0);
    }
#endif
    __syncthreads();

    CFrag sc[2];
    #pragma unroll
    for (int tt = 0; tt < 2; ++tt)
      #pragma unroll
      for (int r = 0; r < 8; ++r) sc[tt].f[r] = 0.0f;

    // scores: S[16 x 32] = Q(16x64) * K^T, K from LDS (contiguous reads)
    #pragma unroll
    for (int tt = 0; tt < 2; ++tt) {
      const int trow = tt * 16 + l15;
      #pragma unroll
      for (int dc = 0; dc < 2; ++dc) {
        AFrag bk;
        const bf16* kr = &Ks[buf][trow][dc * 32 + half * 16];
        #pragma unroll
        for (int i = 0; i < 16; ++i) bk.e[i] = kr[i];
        sc[tt].v = wmma_bf16(aq[dc].v, bk.v, sc[tt].v);
      }
    }

    // online softmax; per-row state per lane (rows = r + 8*half)
    float p0[8], p1[8];
    #pragma unroll
    for (int r = 0; r < 8; ++r) {
      const float x0 = sc[0].f[r] * scale;
      const float x1 = sc[1].f[r] * scale;
      float m = fmaxf(x0, x1);
      #pragma unroll
      for (int off = 8; off > 0; off >>= 1)
        m = fmaxf(m, __shfl_xor(m, off, 16));
      const float mnew = fmaxf(mrun[r], m);
      const float corr = __expf(mrun[r] - mnew);
      p0[r] = __expf(x0 - mnew);
      p1[r] = __expf(x1 - mnew);
      float rs = p0[r] + p1[r];
      #pragma unroll
      for (int off = 8; off > 0; off >>= 1)
        rs += __shfl_xor(rs, off, 16);
      lrun[r] = lrun[r] * corr + rs;
      mrun[r] = mnew;
      #pragma unroll
      for (int dt = 0; dt < 4; ++dt) acc[dt].f[r] *= corr;
    }

    // P: C-layout -> LDS [row][t] -> A-layout (per-wave region)
    #pragma unroll
    for (int r = 0; r < 8; ++r) {
      const int row = r + half * 8;
      Plds[wave][row][l15]      = (bf16)p0[r];
      Plds[wave][row][16 + l15] = (bf16)p1[r];
    }
    AFrag pa;
    {
      const int kb = half * 8;
      #pragma unroll
      for (int i = 0; i < 16; ++i)
        pa.e[i] = Plds[wave][l15][half * 16 + i];
    }

    // ctx += P(16x32) * V(32x64), V from LDS
    #pragma unroll
    for (int dt = 0; dt < 4; ++dt) {
      AFrag bv;
      const int d = dt * 16 + l15;
      #pragma unroll
      for (int i = 0; i < 16; ++i)
        bv.e[i] = Vs[buf][half * 16 + i][d];
      acc[dt].v = wmma_bf16(pa.v, bv.v, acc[dt].v);
    }
    __syncthreads();
  }

  // normalize, store ctx bf16 in [b, s, h*64 + d]
  const int bb = bh >> 4, hh = bh & 15;
  #pragma unroll
  for (int dt = 0; dt < 4; ++dt)
    #pragma unroll
    for (int r = 0; r < 8; ++r) {
      const int srow = qrow0 + r + half * 8;
      const int d    = dt * 16 + l15;
      ctx[((size_t)bb * SEQ + srow) * EMB + hh * HD + d] =
          (bf16)(acc[dt].f[r] / lrun[r]);
    }
}

// ---------------------------------------------------------------------------
// Kernel 3: output projection + residual.  y = dec + ctx(4096x1024 bf16) @ Wo
// ---------------------------------------------------------------------------
__global__ __launch_bounds__(256) void out_proj_kernel(
    const bf16* __restrict__ ctx, const float* __restrict__ Wo,
    const float* __restrict__ dec, float* __restrict__ y)
{
  const int n0 = blockIdx.x * 64;
  const int m0 = blockIdx.y * 128;

  __shared__ bf16 As[128][34];
  __shared__ bf16 Bs[32][66];

  const int tid  = threadIdx.x;
  const int lane = tid & 31;
  const int wave = tid >> 5;
  const int wm = wave & 3, wn = wave >> 2;
  const int half = lane >> 4, l15 = lane & 15;

  CFrag c[2][2];
  #pragma unroll
  for (int mi = 0; mi < 2; ++mi)
    #pragma unroll
    for (int ni = 0; ni < 2; ++ni)
      #pragma unroll
      for (int r = 0; r < 8; ++r) c[mi][ni].f[r] = 0.0f;

  for (int k0 = 0; k0 < EMB; k0 += 32) {
    { // A tile (already bf16)
      const int row = tid >> 1, hc = (tid & 1) * 16;
      const bf16* src = ctx + (size_t)(m0 + row) * EMB + k0 + hc;
      __builtin_prefetch(src + 64, 0, 0);
      #pragma unroll
      for (int j = 0; j < 16; ++j) As[row][hc + j] = src[j];
    }
    { // B tile: Wo[k][n] f32 -> bf16
      const int kr = tid >> 3, cg = (tid & 7) * 8;
      const float4* src = (const float4*)(Wo + (size_t)(k0 + kr) * EMB + n0 + cg);
      #pragma unroll
      for (int j = 0; j < 2; ++j) {
        float4 f = src[j];
        const int base = cg + j * 4;
        Bs[kr][base + 0] = (bf16)f.x;
        Bs[kr][base + 1] = (bf16)f.y;
        Bs[kr][base + 2] = (bf16)f.z;
        Bs[kr][base + 3] = (bf16)f.w;
      }
    }
    __syncthreads();

    AFrag a[2], b[2];
    const int kb = half * 8, ks = half * 16;
    #pragma unroll
    for (int mi = 0; mi < 2; ++mi) {
      const int row = wm * 32 + mi * 16 + l15;
      #pragma unroll
      for (int i = 0; i < 16; ++i)
        a[mi].e[i] = As[row][ks + i];
    }
    #pragma unroll
    for (int ni = 0; ni < 2; ++ni) {
      const int col = wn * 32 + ni * 16 + l15;
      #pragma unroll
      for (int i = 0; i < 16; ++i)
        b[ni].e[i] = Bs[ks + i][col];
    }
    #pragma unroll
    for (int mi = 0; mi < 2; ++mi)
      #pragma unroll
      for (int ni = 0; ni < 2; ++ni)
        c[mi][ni].v = wmma_bf16(a[mi].v, b[ni].v, c[mi][ni].v);
    __syncthreads();
  }

  #pragma unroll
  for (int mi = 0; mi < 2; ++mi)
    #pragma unroll
    for (int ni = 0; ni < 2; ++ni)
      #pragma unroll
      for (int r = 0; r < 8; ++r) {
        const int m = m0 + wm * 32 + mi * 16 + r + half * 8;
        const int n = n0 + wn * 32 + ni * 16 + l15;
        y[(size_t)m * EMB + n] = dec[(size_t)m * EMB + n] + c[mi][ni].f[r];
      }
}

// ---------------------------------------------------------------------------
// Kernel 4: LayerNorm per row of 1024 features.
// ---------------------------------------------------------------------------
__global__ __launch_bounds__(256) void layernorm_kernel(
    const float* __restrict__ y, const float* __restrict__ gamma,
    const float* __restrict__ beta, float* __restrict__ out)
{
  const int row = blockIdx.x;
  const float* yr = y + (size_t)row * EMB;

  float s = 0.0f, sq = 0.0f;
  for (int i = threadIdx.x; i < EMB; i += 256) {
    const float x = yr[i];
    s += x; sq += x * x;
  }
  #pragma unroll
  for (int off = 16; off > 0; off >>= 1) {
    s  += __shfl_xor(s,  off, 32);
    sq += __shfl_xor(sq, off, 32);
  }
  __shared__ float rs[8], rq[8], mu_s, rv_s;
  if ((threadIdx.x & 31) == 0) { rs[threadIdx.x >> 5] = s; rq[threadIdx.x >> 5] = sq; }
  __syncthreads();
  if (threadIdx.x == 0) {
    float ts = 0.0f, tq = 0.0f;
    #pragma unroll
    for (int w = 0; w < 8; ++w) { ts += rs[w]; tq += rq[w]; }
    const float mu  = ts * (1.0f / EMB);
    const float var = tq * (1.0f / EMB) - mu * mu;
    mu_s = mu;
    rv_s = rsqrtf(var + LN_EPSF);
  }
  __syncthreads();
  const float mu = mu_s, rv = rv_s;
  for (int i = threadIdx.x; i < EMB; i += 256)
    out[(size_t)row * EMB + i] = (yr[i] - mu) * rv * gamma[i] + beta[i];
}

// ---------------------------------------------------------------------------
extern "C" void kernel_launch(void* const* d_in, const int* in_sizes, int n_in,
                              void* d_out, int out_size, void* d_ws, size_t ws_size,
                              hipStream_t stream) {
  const float* enc   = (const float*)d_in[0];
  const float* dec   = (const float*)d_in[1];
  const float* Wq    = (const float*)d_in[2];
  const float* Wk    = (const float*)d_in[3];
  const float* Wv    = (const float*)d_in[4];
  const float* Wo    = (const float*)d_in[5];
  const float* gamma = (const float*)d_in[6];
  const float* beta  = (const float*)d_in[7];
  float* out = (float*)d_out;

  char* ws = (char*)d_ws;
  const size_t HB = (size_t)BATCH * NH * SEQ * HD * sizeof(bf16); // 8 MiB
  bf16*  qws = (bf16*)(ws + 0 * HB);
  bf16*  kws = (bf16*)(ws + 1 * HB);
  bf16*  vws = (bf16*)(ws + 2 * HB);
  bf16*  cws = (bf16*)(ws + 3 * HB);          // ctx bf16, 8 MiB
  float* yws = (float*)(ws + 4 * HB);         // pre-LN f32, 16 MiB

  qkv_proj_kernel<<<dim3(EMB / 64, MROWS / 128, 3), 256, 0, stream>>>(
      enc, dec, Wq, Wk, Wv, qws, kws, vws);
  attn_kernel<<<dim3(SEQ / 128, BATCH * NH), 256, 0, stream>>>(
      qws, kws, vws, cws);
  out_proj_kernel<<<dim3(EMB / 64, MROWS / 128), 256, 0, stream>>>(
      cws, Wo, dec, yws);
  layernorm_kernel<<<MROWS, 256, 0, stream>>>(yws, gamma, beta, out);
}
